// MyRNN_6536940224881
// MI455X (gfx1250) — compile-verified, matrix operand order checked by reading, not run on hardware
//
#include <hip/hip_runtime.h>

#define B_ 64
#define S_ 2048
#define E_ 256
#define H_ 256

typedef __attribute__((ext_vector_type(16))) __bf16 v16bf;
typedef __attribute__((ext_vector_type(8)))  float  v8f;

struct U4x2 { uint4 a, b; };

__device__ inline v16bf frag_from_u8(uint4 a, uint4 b) {
    U4x2 s{a, b};
    return __builtin_bit_cast(v16bf, s);
}

// ---- f32 -> packed bf16x2 conversion -------------------------------------
#if __has_builtin(__builtin_amdgcn_cvt_pk_bf16_f32)
typedef __attribute__((ext_vector_type(2))) __bf16 v2bf;
__device__ inline unsigned pack2(float lo, float hi) {
    v2bf r = __builtin_amdgcn_cvt_pk_bf16_f32(lo, hi);
    return __builtin_bit_cast(unsigned, r);
}
__device__ inline unsigned short f2bf16(float f) {
    return (unsigned short)(pack2(f, f) & 0xFFFFu);
}
#elif __has_builtin(__builtin_amdgcn_perm)
// round-to-nearest (ties away) + single v_perm_b32 pack: 3 VALU per pair
__device__ inline unsigned pack2(float lo, float hi) {
    unsigned ul = __builtin_bit_cast(unsigned, lo) + 0x8000u;
    unsigned uh = __builtin_bit_cast(unsigned, hi) + 0x8000u;
    // result bytes: [1:0] = ul bytes[3:2], [3:2] = uh bytes[3:2]
    return __builtin_amdgcn_perm(uh, ul, 0x07060302u);
}
__device__ inline unsigned short f2bf16(float f) {
    return (unsigned short)((__builtin_bit_cast(unsigned, f) + 0x8000u) >> 16);
}
#else
// fallback: round-to-nearest-even bit math
__device__ inline unsigned f2bf_(float f) {
    unsigned u = __builtin_bit_cast(unsigned, f);
    return (u + 0x7FFFu + ((u >> 16) & 1u)) >> 16;
}
__device__ inline unsigned pack2(float lo, float hi) {
    return f2bf_(lo) | (f2bf_(hi) << 16);
}
__device__ inline unsigned short f2bf16(float f) {
    return (unsigned short)f2bf_(f);
}
#endif

// ---------------------------------------------------------------------------
// Phase 1: xw[B*S, H] = x[B*S, E] @ Ww^T + Wb, written into d_out (overwritten
// later by the scan with h3). WG = 256 threads (8 waves), tile M=128, N=256.
// Ww staged in LDS as bf16 (64KB per K-half), XOR-swizzled against bank
// conflicts on the 16-rows-same-offset B-fragment reads.
// ---------------------------------------------------------------------------
__global__ __launch_bounds__(256)
void rnn_xw_gemm(const float* __restrict__ x, const float* __restrict__ Ww,
                 const float* __restrict__ Wb, float* __restrict__ out) {
    __shared__ __align__(16) unsigned sB[H_ * 64]; // [n][64 dw] = bf16[n][128], 64KB

    const int tid     = threadIdx.x;
    const int lane    = tid & 31;
    const int wave    = tid >> 5;
    const int halfsel = lane >> 4;   // 0: lanes 0-15, 1: lanes 16-31
    const int l16     = lane & 15;
    const long r0     = (long)blockIdx.x * 128 + wave * 16; // wave's M-strip base row

    v8f acc[16];
    #pragma unroll
    for (int nt = 0; nt < 16; ++nt) acc[nt] = v8f{};

    for (int kc = 0; kc < 2; ++kc) {
        // ---- stage Ww[:, kc*128 .. +127] as bf16 into LDS (thread t = row t) ----
        {
            const int n = tid;
            const float* src = Ww + (long)n * E_ + kc * 128;
            #pragma unroll
            for (int g = 0; g < 16; ++g) {      // 16 groups of 4 dwords (8 f32)
                float4 f0 = ((const float4*)src)[g * 2 + 0];
                float4 f1 = ((const float4*)src)[g * 2 + 1];
                uint4 u;
                u.x = pack2(f0.x, f0.y); u.y = pack2(f0.z, f0.w);
                u.z = pack2(f1.x, f1.y); u.w = pack2(f1.z, f1.w);
                ((uint4*)sB)[n * 16 + (g ^ (n & 15))] = u;   // XOR swizzle
            }
        }
        __syncthreads();

        #pragma unroll
        for (int kb = 0; kb < 4; ++kb) {
            // ---- A fragment from global x (rows r0+l16, K block) ----
            const float* arow = x + (r0 + l16) * (long)E_ + kc * 128 + kb * 32 + halfsel * 8;
            float4 a0 = ((const float4*)arow)[0];
            float4 a1 = ((const float4*)arow)[1];
            float4 a2 = ((const float4*)(arow + 16))[0];
            float4 a3 = ((const float4*)(arow + 16))[1];
            uint4 pa, pb;
            pa.x = pack2(a0.x, a0.y); pa.y = pack2(a0.z, a0.w);
            pa.z = pack2(a1.x, a1.y); pa.w = pack2(a1.z, a1.w);
            pb.x = pack2(a2.x, a2.y); pb.y = pack2(a2.z, a2.w);
            pb.z = pack2(a3.x, a3.y); pb.w = pack2(a3.z, a3.w);
            v16bf afrag = frag_from_u8(pa, pb);

            const int g1 = (kb * 4 + halfsel) ^ l16;
            const int g2 = (kb * 4 + halfsel + 2) ^ l16;
            #pragma unroll
            for (int nt = 0; nt < 16; ++nt) {
                const int rowbase = (nt * 16 + l16) * 16;
                uint4 b0 = ((const uint4*)sB)[rowbase + g1];
                uint4 b1 = ((const uint4*)sB)[rowbase + g2];
                v16bf bfrag = frag_from_u8(b0, b1);
                acc[nt] = __builtin_amdgcn_wmma_f32_16x16x32_bf16(
                    false, afrag, false, bfrag, (short)0, acc[nt], false, false);
            }
        }
        __syncthreads();
    }

    // ---- epilogue: + Wb, store f32 ----
    #pragma unroll
    for (int nt = 0; nt < 16; ++nt) {
        const int n = nt * 16 + l16;
        const float wb = Wb[n];
        #pragma unroll
        for (int i = 0; i < 8; ++i) {
            const long r = r0 + i + halfsel * 8;        // C layout: VGPR i -> M=i / i+8
            out[r * H_ + n] = acc[nt][i] + wb;
        }
    }
}

// ---------------------------------------------------------------------------
// Phase 2: sequential scan. 4 WGs x 16 batch rows, 512 threads = 16 waves,
// wave nt owns output columns [nt*16, nt*16+16). Uw fragments register-
// resident; h double-buffered in LDS (XOR swizzled); xw prefetched 1 step.
// ---------------------------------------------------------------------------
__global__ __launch_bounds__(512)
void rnn_scan(const float* __restrict__ Uw, const float* __restrict__ Ub,
              float* __restrict__ out) {
    __shared__ __align__(16) unsigned hbuf[2][16 * 128]; // bf16[16][256] x2 = 16KB

    const int tid     = threadIdx.x;
    const int lane    = tid & 31;
    const int nt      = tid >> 5;            // wave id = N tile
    const int halfsel = lane >> 4;
    const int l16     = lane & 15;
    const int b0      = blockIdx.x * 16;

    const int n = nt * 16 + l16;             // lane's output column (fixed)

    // ---- register-resident B fragments: B[k][n] = Uw[n][k] ----
    v16bf Bfrag[8];
    #pragma unroll
    for (int kb = 0; kb < 8; ++kb) {
        const float* urow = Uw + (long)n * H_ + kb * 32 + halfsel * 8;
        float4 c0 = ((const float4*)urow)[0];
        float4 c1 = ((const float4*)urow)[1];
        float4 c2 = ((const float4*)(urow + 16))[0];
        float4 c3 = ((const float4*)(urow + 16))[1];
        uint4 pa, pb;
        pa.x = pack2(c0.x, c0.y); pa.y = pack2(c0.z, c0.w);
        pa.z = pack2(c1.x, c1.y); pa.w = pack2(c1.z, c1.w);
        pb.x = pack2(c2.x, c2.y); pb.y = pack2(c2.z, c2.w);
        pb.z = pack2(c3.x, c3.y); pb.w = pack2(c3.z, c3.w);
        Bfrag[kb] = frag_from_u8(pa, pb);
    }
    const float ub = Ub[n];

    v8f acc = v8f{};                         // t (carry), starts at 0

    // prefetch xw for s=0; element i -> row m = i + halfsel*8
    float xw_cur[8];
    #pragma unroll
    for (int i = 0; i < 8; ++i)
        xw_cur[i] = out[(long)(b0 + i + halfsel * 8) * S_ * H_ + n];

    int buf = 0;
    for (int s = 0; s < S_; ++s) {
        // h = tanh(xw + t): write f32 h to out (in place), bf16 h to LDS
        #pragma unroll
        for (int i = 0; i < 8; ++i) {
            const int m = i + halfsel * 8;
            const float h = tanhf(xw_cur[i] + acc[i]);
            out[((long)(b0 + m) * S_ + s) * H_ + n] = h;
            // swizzled bf16 store at (m, n)
            const int sdw = (((n >> 3) ^ m) << 2) | ((n >> 1) & 3);
            ((unsigned short*)hbuf[buf])[(m * 128 + sdw) * 2 + (n & 1)] = f2bf16(h);
        }

        // prefetch next step's xw while WMMAs run
        float xw_next[8] = {0, 0, 0, 0, 0, 0, 0, 0};
        if (s + 1 < S_) {
            #pragma unroll
            for (int i = 0; i < 8; ++i)
                xw_next[i] =
                    out[((long)(b0 + i + halfsel * 8) * S_ + (s + 1)) * H_ + n];
        }

        __syncthreads();

        // t_new = h @ Uw^T + Ub
        v8f a2 = v8f{};
        #pragma unroll
        for (int kb = 0; kb < 8; ++kb) {
            const uint4* hb = (const uint4*)hbuf[buf];
            uint4 h0 = hb[l16 * 32 + ((kb * 4 + halfsel) ^ l16)];
            uint4 h1 = hb[l16 * 32 + ((kb * 4 + halfsel + 2) ^ l16)];
            v16bf afrag = frag_from_u8(h0, h1);
            a2 = __builtin_amdgcn_wmma_f32_16x16x32_bf16(
                false, afrag, false, Bfrag[kb], (short)0, a2, false, false);
        }
        #pragma unroll
        for (int i = 0; i < 8; ++i) acc[i] = a2[i] + ub;
        #pragma unroll
        for (int i = 0; i < 8; ++i) xw_cur[i] = xw_next[i];
        buf ^= 1;
    }

    // t_final at d_out + B*S*H
    #pragma unroll
    for (int i = 0; i < 8; ++i) {
        const int m = i + halfsel * 8;
        out[(long)B_ * S_ * H_ + (long)(b0 + m) * H_ + n] = acc[i];
    }
}

extern "C" void kernel_launch(void* const* d_in, const int* in_sizes, int n_in,
                              void* d_out, int out_size, void* d_ws, size_t ws_size,
                              hipStream_t stream) {
    (void)in_sizes; (void)n_in; (void)d_ws; (void)ws_size; (void)out_size;
    const float* x  = (const float*)d_in[0];
    const float* Ww = (const float*)d_in[1];
    const float* Wb = (const float*)d_in[2];
    const float* Uw = (const float*)d_in[3];
    const float* Ub = (const float*)d_in[4];
    float* out = (float*)d_out;

    rnn_xw_gemm<<<dim3((B_ * S_) / 128), dim3(256), 0, stream>>>(x, Ww, Wb, out);
    rnn_scan<<<dim3(B_ / 16), dim3(512), 0, stream>>>(Uw, Ub, out);
}
